// HypergraphNN_26061861552225
// MI455X (gfx1250) — matrix operand order repaired and checked by reading, not compiled, not run on hardware
//
#include <hip/hip_runtime.h>

#define N_NODES 16384
#define N_EDGES 8192
#define D_FEAT  64

typedef __bf16 bf16;
typedef bf16  bf16x4  __attribute__((ext_vector_type(4)));
typedef bf16  bf16x8  __attribute__((ext_vector_type(8)));
typedef bf16  bf16x16 __attribute__((ext_vector_type(16)));
typedef float floatx8 __attribute__((ext_vector_type(8)));

// LDS tile row stride (elements): 32 K-values + 8 pad -> 80B rows, 16B-aligned.
#define LDSP 40

// ---------------------------------------------------------------------------
// Kernel 1: d_v[n] = sum_m H[n, m]   (one wave per row, coalesced float4)
// ---------------------------------------------------------------------------
__global__ __launch_bounds__(256) void degree_rows(const float* __restrict__ H,
                                                   float* __restrict__ dv) {
    const int wave = threadIdx.x >> 5;
    const int lane = threadIdx.x & 31;
    const int row  = blockIdx.x * 8 + wave;
    const float4* p = reinterpret_cast<const float4*>(H + (size_t)row * N_EDGES) + lane;
    float s0 = 0.f, s1 = 0.f;
#pragma unroll 8
    for (int i = 0; i < N_EDGES / 128; ++i) {
        float4 v = p[(size_t)i * 32];
        s0 += v.x + v.y;
        s1 += v.z + v.w;
    }
    float s = s0 + s1;
#pragma unroll
    for (int off = 16; off > 0; off >>= 1) s += __shfl_xor(s, off, 32);
    if (lane == 0) dv[row] = s;
}

// ---------------------------------------------------------------------------
// Kernel 2: Ys[m, d] = d_e[m] * sum_n H[n, m] * d_v[n] * X[n, d]
//   64 threads (2 waves) own 32 m-rows, full K = N_NODES loop, 256 blocks.
//   Cross-barrier software pipeline: tile it+1's global loads are issued
//   before the barrier of iteration it, so they fly across the barrier and
//   the whole next WMMA phase. LDS double-buffered (one barrier / K-step).
//   d_e[m] accumulated in fp32 during staging, applied before the store.
// ---------------------------------------------------------------------------
__global__ __launch_bounds__(64) void gemm1(const float* __restrict__ H,
                                            const float* __restrict__ X,
                                            const float* __restrict__ dv,
                                            float* __restrict__ Ys) {
    __shared__ __align__(16) bf16 HtBuf[2][32 * LDSP];  // [m_local][n_local] (H^T tile)
    __shared__ __align__(16) bf16 XtBuf[2][64 * LDSP];  // [d]      [n_local] (dv-scaled X^T)
    __shared__ float deLds[32];

    const int t    = threadIdx.x;
    const int lane = t & 31;
    const int w    = t >> 5;          // wave 0..1
    const int m0   = blockIdx.x * 32;

    // H staging: K-rows 4*r4h..+3, m-cols ccH..ccH+3
    const int r4h = t >> 3;           // 0..7
    const int ccH = (t & 7) * 4;      // 0..28
    // X staging: K-rows 8*r8x..+7, d-cols ccX..ccX+3
    const int r8x = t >> 4;           // 0..3
    const int ccX = (t & 15) * 4;     // 0..60

    // WMMA fragment mapping (wave32)
    const int half = lane >> 4;
    const int mn   = lane & 15;

    floatx8 acc[4] = {};
    float de4[4] = {0.f, 0.f, 0.f, 0.f};
    float hv[4][4], xv[8][4];

    if (t < 32) deLds[t] = 0.f;

    auto loadTiles = [&](int k0) {
#pragma unroll
        for (int j = 0; j < 4; ++j) {
            float4 q = *reinterpret_cast<const float4*>(
                H + (size_t)(k0 + 4 * r4h + j) * N_EDGES + m0 + ccH);
            hv[j][0] = q.x; hv[j][1] = q.y; hv[j][2] = q.z; hv[j][3] = q.w;
        }
#pragma unroll
        for (int j = 0; j < 8; ++j) {
            const int row = 8 * r8x + j;
            const float s = dv[k0 + row];
            float4 p = *reinterpret_cast<const float4*>(
                X + (size_t)(k0 + row) * D_FEAT + ccX);
            xv[j][0] = p.x * s; xv[j][1] = p.y * s;
            xv[j][2] = p.z * s; xv[j][3] = p.w * s;
        }
    };
    auto storeTiles = [&](bf16* Ht, bf16* Xt) {
#pragma unroll
        for (int i = 0; i < 4; ++i) {
            de4[i] += (hv[0][i] + hv[1][i]) + (hv[2][i] + hv[3][i]);
            bf16x4 ph = { (bf16)hv[0][i], (bf16)hv[1][i],
                          (bf16)hv[2][i], (bf16)hv[3][i] };
            *reinterpret_cast<bf16x4*>(&Ht[(ccH + i) * LDSP + 4 * r4h]) = ph;
            bf16x8 px = { (bf16)xv[0][i], (bf16)xv[1][i], (bf16)xv[2][i], (bf16)xv[3][i],
                          (bf16)xv[4][i], (bf16)xv[5][i], (bf16)xv[6][i], (bf16)xv[7][i] };
            *reinterpret_cast<bf16x8*>(&Xt[(ccX + i) * LDSP + 8 * r8x]) = px;
        }
    };
    auto compute = [&](const bf16* Ht, const bf16* Xt) {
        bf16x16 a;
        {
            const int mA = 16 * w + mn;
            bf16x8 lo = *reinterpret_cast<const bf16x8*>(&Ht[mA * LDSP + 8 * half]);
            bf16x8 hi = *reinterpret_cast<const bf16x8*>(&Ht[mA * LDSP + 16 + 8 * half]);
#pragma unroll
            for (int i = 0; i < 8; ++i) { a[i] = lo[i]; a[8 + i] = hi[i]; }
        }
#pragma unroll
        for (int tb = 0; tb < 4; ++tb) {
            const int d = tb * 16 + mn;
            bf16x16 b;
            bf16x8 lo = *reinterpret_cast<const bf16x8*>(&Xt[d * LDSP + 8 * half]);
            bf16x8 hi = *reinterpret_cast<const bf16x8*>(&Xt[d * LDSP + 16 + 8 * half]);
#pragma unroll
            for (int i = 0; i < 8; ++i) { b[i] = lo[i]; b[8 + i] = hi[i]; }
            acc[tb] = __builtin_amdgcn_wmma_f32_16x16x32_bf16(
                false, a, false, b, (short)0, acc[tb], false, false);
        }
    };

    bf16* HtCur = HtBuf[0];  bf16* HtNxt = HtBuf[1];
    bf16* XtCur = XtBuf[0];  bf16* XtNxt = XtBuf[1];

    constexpr int NIT = N_NODES / 32;
    loadTiles(0);
    storeTiles(HtCur, XtCur);   // stage tile 0
    loadTiles(32);              // tile 1 loads in flight across the barrier
    __syncthreads();

    for (int it = 1; it < NIT - 1; ++it) {
        compute(HtCur, XtCur);        // WMMAs on tile it-1 (loads for tile it fly under)
        storeTiles(HtNxt, XtNxt);     // stage tile it (its loads landed long ago)
        loadTiles((it + 1) * 32);     // issue tile it+1 BEFORE the fence
        __syncthreads();
        bf16* tp;
        tp = HtCur; HtCur = HtNxt; HtNxt = tp;
        tp = XtCur; XtCur = XtNxt; XtNxt = tp;
    }
    compute(HtCur, XtCur);            // tile NIT-2
    storeTiles(HtNxt, XtNxt);         // stage tile NIT-1
    __syncthreads();
    compute(HtNxt, XtNxt);            // tile NIT-1

    // ---- fold d_e partials (thread always owned columns ccH..ccH+3)
    __syncthreads();
#pragma unroll
    for (int i = 0; i < 4; ++i) atomicAdd(&deLds[ccH + i], de4[i]);
    __syncthreads();

    // ---- scale by d_e and store Ys
#pragma unroll
    for (int r = 0; r < 8; ++r) {
        const int   ml = 16 * w + r + 8 * half;
        const float de = deLds[ml];
        const size_t row = (size_t)(m0 + ml) * D_FEAT;
#pragma unroll
        for (int tb = 0; tb < 4; ++tb)
            Ys[row + tb * 16 + mn] = acc[tb][r] * de;
    }
}

// ---------------------------------------------------------------------------
// Kernel 3: out[n, d] = d_v[n] * sum_m H[n, m] * Ys[m, d]
//   64 threads (2 waves) own 32 n-rows, 512 blocks. K = m is row-contiguous
//   in H -> A fragments stream from global (b128), register-prefetched one
//   K-step ahead across the barrier. Ys staged via double-buffered LDS.
// ---------------------------------------------------------------------------
__global__ __launch_bounds__(64) void gemm2(const float* __restrict__ H,
                                            const float* __restrict__ Ys,
                                            const float* __restrict__ dv,
                                            float* __restrict__ out) {
    __shared__ __align__(16) bf16 YtBuf[2][64 * LDSP];  // [d][k_local]

    const int t    = threadIdx.x;
    const int lane = t & 31;
    const int w    = t >> 5;
    const int n0   = blockIdx.x * 32;

    const int r8 = t >> 4;            // 0..3
    const int cc = (t & 15) * 4;      // 0..60

    const int half = lane >> 4;
    const int mn   = lane & 15;

    const float* Hrow = H + (size_t)(n0 + 16 * w + mn) * N_EDGES;

    floatx8 acc[4] = {};
    float yv[8][4];
    float4 av[4], av2[4];             // A fp32: K runs {8h..+7, 16+8h..+7}

    auto loadYs = [&](int k0) {
#pragma unroll
        for (int j = 0; j < 8; ++j) {
            float4 q = *reinterpret_cast<const float4*>(
                Ys + (size_t)(k0 + 8 * r8 + j) * D_FEAT + cc);
            yv[j][0] = q.x; yv[j][1] = q.y; yv[j][2] = q.z; yv[j][3] = q.w;
        }
    };
    auto storeYs = [&](bf16* Yt) {
#pragma unroll
        for (int i = 0; i < 4; ++i) {
            bf16x8 pk = { (bf16)yv[0][i], (bf16)yv[1][i], (bf16)yv[2][i], (bf16)yv[3][i],
                          (bf16)yv[4][i], (bf16)yv[5][i], (bf16)yv[6][i], (bf16)yv[7][i] };
            *reinterpret_cast<bf16x8*>(&Yt[(cc + i) * LDSP + 8 * r8]) = pk;
        }
    };
    auto loadA = [&](int k0, float4* a4) {
        const float* p1 = Hrow + k0 + 8 * half;
        const float* p2 = Hrow + k0 + 16 + 8 * half;
        a4[0] = *reinterpret_cast<const float4*>(p1);
        a4[1] = *reinterpret_cast<const float4*>(p1 + 4);
        a4[2] = *reinterpret_cast<const float4*>(p2);
        a4[3] = *reinterpret_cast<const float4*>(p2 + 4);
    };
    auto cvtA = [&]() {
        bf16x16 a;
        a[0]  = (bf16)av[0].x; a[1]  = (bf16)av[0].y; a[2]  = (bf16)av[0].z; a[3]  = (bf16)av[0].w;
        a[4]  = (bf16)av[1].x; a[5]  = (bf16)av[1].y; a[6]  = (bf16)av[1].z; a[7]  = (bf16)av[1].w;
        a[8]  = (bf16)av[2].x; a[9]  = (bf16)av[2].y; a[10] = (bf16)av[2].z; a[11] = (bf16)av[2].w;
        a[12] = (bf16)av[3].x; a[13] = (bf16)av[3].y; a[14] = (bf16)av[3].z; a[15] = (bf16)av[3].w;
        return a;
    };
    auto computeB = [&](const bf16* Yt, const bf16x16& a) {
#pragma unroll
        for (int tb = 0; tb < 4; ++tb) {
            const int d = tb * 16 + mn;
            bf16x16 b;
            bf16x8 lo = *reinterpret_cast<const bf16x8*>(&Yt[d * LDSP + 8 * half]);
            bf16x8 hi = *reinterpret_cast<const bf16x8*>(&Yt[d * LDSP + 16 + 8 * half]);
#pragma unroll
            for (int i = 0; i < 8; ++i) { b[i] = lo[i]; b[8 + i] = hi[i]; }
            acc[tb] = __builtin_amdgcn_wmma_f32_16x16x32_bf16(
                false, a, false, b, (short)0, acc[tb], false, false);
        }
    };

    bf16* YtCur = YtBuf[0];  bf16* YtNxt = YtBuf[1];

    constexpr int NIT = N_EDGES / 32;
    loadYs(0);
    storeYs(YtCur);                   // stage tile 0
    loadA(0, av);                     // A(0)
    loadYs(32);                       // Ys(1) in flight across barrier
    loadA(32, av2);                   // A(1)  in flight across barrier
    __syncthreads();

    for (int it = 1; it < NIT - 1; ++it) {
        bf16x16 a = cvtA();           // A(it-1), ready
        computeB(YtCur, a);           // tile it-1; next-tile loads fly under
        storeYs(YtNxt);               // stage Ys(it)
#pragma unroll
        for (int j = 0; j < 4; ++j) av[j] = av2[j];   // A(it), long landed
        loadYs((it + 1) * 32);        // issue tile it+1 BEFORE the fence
        loadA((it + 1) * 32, av2);
        __syncthreads();
        bf16* tp = YtCur; YtCur = YtNxt; YtNxt = tp;
    }
    {
        bf16x16 a = cvtA();
        computeB(YtCur, a);           // tile NIT-2
        storeYs(YtNxt);               // stage tile NIT-1
#pragma unroll
        for (int j = 0; j < 4; ++j) av[j] = av2[j];
        __syncthreads();
        bf16x16 a2 = cvtA();
        computeB(YtNxt, a2);          // tile NIT-1
    }

    // ---- scale by d_v and store
#pragma unroll
    for (int r = 0; r < 8; ++r) {
        const int   nl  = 16 * w + r + 8 * half;
        const float dvn = dv[n0 + nl];
        const size_t row = (size_t)(n0 + nl) * D_FEAT;
#pragma unroll
        for (int tb = 0; tb < 4; ++tb)
            out[row + tb * 16 + mn] = acc[tb][r] * dvn;
    }
}

// ---------------------------------------------------------------------------
extern "C" void kernel_launch(void* const* d_in, const int* in_sizes, int n_in,
                              void* d_out, int out_size, void* d_ws, size_t ws_size,
                              hipStream_t stream) {
    const float* H = (const float*)d_in[0];   // [N_NODES, N_EDGES] fp32
    const float* X = (const float*)d_in[1];   // [N_NODES, D_FEAT]  fp32
    float* out = (float*)d_out;               // [N_NODES, D_FEAT]  fp32

    // workspace: d_v (N_NODES floats) | Ys (N_EDGES*D_FEAT floats) ~= 2.1 MB
    float* dv = (float*)d_ws;
    float* Ys = dv + N_NODES;

    degree_rows<<<N_NODES / 8, 256, 0, stream>>>(H, dv);
    gemm1<<<N_EDGES / 32, 64, 0, stream>>>(H, X, dv, Ys);
    gemm2<<<N_NODES / 32, 64, 0, stream>>>(H, Ys, dv, out);
}
